// MpnnGGNN_45045617001093
// MI455X (gfx1250) — matrix operand order repaired and checked by reading, not compiled, not run on hardware
//
#include <hip/hip_runtime.h>

// MPNN-GGNN for MI455X (gfx1250, wave32).  All heavy GEMMs use
// v_wmma_f32_16x16x32_bf16 (bf16 inputs, f32 accumulate).
// k_gemm computes a 16x64 tile per wave (1 A-frag reused over 4 B-frags,
// ~12.8 flop/byte) and supports a deterministic K-split for the K=16384
// message GEMM (partials + fixed-order reduce; no atomics).
// Workspace use: ~112 MB (f stored bf16: 67MB, G^T: 33.5MB, partials 4MB).

typedef __attribute__((ext_vector_type(16))) __bf16 v16bf;
typedef __attribute__((ext_vector_type(8)))  float  v8f;
typedef __attribute__((ext_vector_type(4)))  unsigned int v4u;

union FragBF { v16bf v; v4u u[2]; };

__device__ inline __bf16 f2bf(float f) {
  unsigned u = __float_as_uint(f);
  unsigned r = (u + 0x7FFFu + ((u >> 16) & 1u)) >> 16;
  unsigned short s = (unsigned short)r;
  return __builtin_bit_cast(__bf16, s);
}
__device__ inline float bf2f(__bf16 h) {
  unsigned short s = __builtin_bit_cast(unsigned short, h);
  return __uint_as_float(((unsigned)s) << 16);
}
__device__ inline v8f zero8() {
  v8f z = {0.f, 0.f, 0.f, 0.f, 0.f, 0.f, 0.f, 0.f};
  return z;
}

// ---------------------------------------------------------------- prep kernels
__global__ void k_convert(const float* __restrict__ s, __bf16* __restrict__ d, int n) {
  int i = blockIdx.x * 256 + threadIdx.x;
  if (i < n) d[i] = f2bf(s[i]);
}

// src [K][N] f32  ->  dst [N][Kpad] bf16 (zero-padded K)
__global__ void k_transpose(const float* __restrict__ src, __bf16* __restrict__ dst,
                            int K, int N, int Kpad) {
  int i = blockIdx.x * 256 + threadIdx.x;
  int total = N * Kpad;
  if (i >= total) return;
  int n = i / Kpad, k = i - n * Kpad;
  dst[i] = (k < K) ? f2bf(src[k * N + n]) : f2bf(0.f);
}

// h0 = pad(h_in, 32 -> 64), written f32 + bf16
__global__ void k_h0(const float* __restrict__ h_in, float* __restrict__ h,
                     __bf16* __restrict__ hb) {
  int i = blockIdx.x * 256 + threadIdx.x;
  if (i >= 2048 * 64) return;
  int row = i >> 6, c = i & 63;
  float v = (c < 32) ? h_in[row * 32 + c] : 0.f;
  h[i] = v;
  hb[i] = f2bf(v);
}

// xg = bf16(concat(h, h_in)) padded to 128 cols
__global__ void k_xg(const __bf16* __restrict__ hb, const float* __restrict__ h_in,
                     __bf16* __restrict__ xg) {
  int i = blockIdx.x * 256 + threadIdx.x;
  if (i >= 2048 * 128) return;
  int row = i >> 7, c = i & 127;
  float v = (c < 64) ? bf2f(hb[row * 64 + c])
                     : ((c < 96) ? h_in[row * 32 + (c - 64)] : 0.f);
  xg[i] = f2bf(v);
}

// fixed-order K-split reduction (deterministic, no atomics)
__global__ void k_reduce(const float* __restrict__ part, float* __restrict__ out,
                         int n, int z) {
  int i = blockIdx.x * 256 + threadIdx.x;
  if (i >= n) return;
  float s = 0.f;
  for (int j = 0; j < z; ++j) s += part[(long)j * n + i];
  out[i] = s;
}

// ------------------------------------------------------- fused edge-network MLP
// rows = am rows (B*N*N = 262144), 64 rows per block, 4 waves.
#define EROWS 64
__global__ void __launch_bounds__(128) k_edge(
    const float* __restrict__ am,
    const float* __restrict__ We0, const float* __restrict__ be0,
    const __bf16* __restrict__ We1T, const float* __restrict__ be1,
    const __bf16* __restrict__ We2T, const float* __restrict__ be2,
    __bf16* __restrict__ f_out) {
  __shared__ float s_am[EROWS * 4];
  __shared__ __align__(16) __bf16 a1[EROWS * 128];
  __shared__ __align__(16) __bf16 a2[EROWS * 256];
  const int tid = threadIdx.x;
  const long row0 = (long)blockIdx.x * EROWS;

  for (int i = tid; i < EROWS * 4; i += 128) s_am[i] = am[row0 * 4 + i];
  __syncthreads();

  // layer 1: K=4, scalar
  for (int i = tid; i < EROWS * 128; i += 128) {
    int r = i >> 7, c = i & 127;
    float acc = be0[c];
#pragma unroll
    for (int e = 0; e < 4; ++e) acc += s_am[r * 4 + e] * We0[e * 128 + c];
    a1[i] = f2bf(fmaxf(acc, 0.f));
  }
  __syncthreads();

  const int wave = tid >> 5, lane = tid & 31;
  const int m = lane & 15, hi = lane >> 4;
  const int wr0 = wave * 16;

  // layer 2: [64,128] @ W[128,256], WMMA
  for (int nt = 0; nt < 16; ++nt) {
    v8f acc = zero8();
    const __bf16* arow = a1 + (wr0 + m) * 128 + hi * 8;
    const __bf16* brow = We1T + (nt * 16 + m) * 128 + hi * 8;
#pragma unroll
    for (int kk = 0; kk < 128; kk += 32) {
      FragBF a, b;
      a.u[0] = *(const v4u*)(arow + kk);
      a.u[1] = *(const v4u*)(arow + kk + 16);
      b.u[0] = *(const v4u*)(brow + kk);
      b.u[1] = *(const v4u*)(brow + kk + 16);
      acc = __builtin_amdgcn_wmma_f32_16x16x32_bf16(false, a.v, false, b.v,
                                                    (short)0, acc, false, false);
    }
#pragma unroll
    for (int r = 0; r < 8; ++r) {
      int orow = wr0 + hi * 8 + r, ocol = nt * 16 + m;
      a2[orow * 256 + ocol] = f2bf(fmaxf(acc[r] + be1[ocol], 0.f));
    }
  }
  __syncthreads();

  // layer 3: [64,256] @ W[256,128], WMMA, write f (bf16) to global
  for (int nt = 0; nt < 8; ++nt) {
    v8f acc = zero8();
    const __bf16* arow = a2 + (wr0 + m) * 256 + hi * 8;
    const __bf16* brow = We2T + (nt * 16 + m) * 256 + hi * 8;
#pragma unroll
    for (int kk = 0; kk < 256; kk += 32) {
      FragBF a, b;
      a.u[0] = *(const v4u*)(arow + kk);
      a.u[1] = *(const v4u*)(arow + kk + 16);
      b.u[0] = *(const v4u*)(brow + kk);
      b.u[1] = *(const v4u*)(brow + kk + 16);
      acc = __builtin_amdgcn_wmma_f32_16x16x32_bf16(false, a.v, false, b.v,
                                                    (short)0, acc, false, false);
    }
#pragma unroll
    for (int r = 0; r < 8; ++r) {
      int orow = wr0 + hi * 8 + r, ocol = nt * 16 + m;
      f_out[(row0 + orow) * 128 + ocol] = f2bf(fmaxf(acc[r] + be2[ocol], 0.f));
    }
  }
}

// --------------------------------------------------------- generic WMMA GEMM
// One wave per block; wave computes a 16 x 64 output tile (4 accumulators,
// A-fragment loaded once per K step and reused over 4 B-fragments).
// A: [rows][lda] bf16, WT: [cols][ldw] bf16 (W transposed).
// K-split over blockIdx.z: chunk z covers K in [z*Kc, min(K,(z+1)*Kc));
// when gridDim.z > 1, mode must be 0 with bias=null/act=0 and each chunk
// writes its own partial plane of rows*ldo floats (reduced by k_reduce).
// mode: 0 = f32 out[row*ldo+col]; 1 = bf16 same; 2 = bf16 G^T scatter
//       (row=b*128+w, col=k*64+o  ->  out[(b*64+o)*16384 + w*128 + k]).
// act: 0 none, 1 relu, 2 sigmoid.
__global__ void __launch_bounds__(32) k_gemm(
    const __bf16* __restrict__ A, long a_bstride, int lda,
    const __bf16* __restrict__ WT, long w_bstride, int ldw,
    const float* __restrict__ bias,
    int K, int Kc, int rows_per_batch, int cols, int act, int mode,
    void* __restrict__ out, int ldo) {
  const int row0 = blockIdx.x * 16;
  const int col0 = blockIdx.y * 64;
  const int kz = blockIdx.z;
  const int k_beg = kz * Kc;
  int k_end = k_beg + Kc;
  if (k_end > K) k_end = K;
  const int ctn = min(4, (cols - col0) >> 4);
  const int batch = row0 / rows_per_batch;
  const int lrow = row0 - batch * rows_per_batch;
  const int lane = threadIdx.x;
  const int m = lane & 15, hi = lane >> 4;

  const __bf16* ap = A + (long)batch * a_bstride + (long)(lrow + m) * lda + hi * 8;
  const __bf16* bp0 = WT + (long)batch * w_bstride + (long)(col0 + m) * ldw + hi * 8;

  v8f acc[4] = {zero8(), zero8(), zero8(), zero8()};
  for (int kk = k_beg; kk < k_end; kk += 32) {
    FragBF a;
    a.u[0] = *(const v4u*)(ap + kk);
    a.u[1] = *(const v4u*)(ap + kk + 16);
    if (Kc >= 2048 && kk + 512 < k_end) {
      __builtin_prefetch(ap + kk + 512, 0, 0);
      __builtin_prefetch(bp0 + kk + 512, 0, 0);
    }
#pragma unroll
    for (int ct = 0; ct < 4; ++ct) {
      if (ct < ctn) {
        FragBF b;
        const __bf16* bp = bp0 + (long)ct * 16 * ldw;
        b.u[0] = *(const v4u*)(bp + kk);
        b.u[1] = *(const v4u*)(bp + kk + 16);
        acc[ct] = __builtin_amdgcn_wmma_f32_16x16x32_bf16(
            false, a.v, false, b.v, (short)0, acc[ct], false, false);
      }
    }
  }

  const long zoff = (long)kz * ((long)gridDim.x * 16) * ldo;  // partial plane
#pragma unroll
  for (int ct = 0; ct < 4; ++ct) {
    if (ct >= ctn) continue;
#pragma unroll
    for (int r = 0; r < 8; ++r) {
      int grow = row0 + hi * 8 + r;
      int gcol = col0 + ct * 16 + m;
      float x = acc[ct][r];
      if (bias) x += bias[gcol];
      if (act == 1) x = fmaxf(x, 0.f);
      else if (act == 2) x = 1.f / (1.f + __expf(-x));
      if (mode == 0) {
        ((float*)out)[zoff + (long)grow * ldo + gcol] = x;
      } else if (mode == 1) {
        ((__bf16*)out)[(long)grow * ldo + gcol] = f2bf(x);
      } else {
        int bb = grow >> 7, w = grow & 127, k = gcol >> 6, o = gcol & 63;
        ((__bf16*)out)[((long)bb * 64 + o) * 16384 + (long)w * 128 + k] = f2bf(x);
      }
    }
  }
}

// -------------------------------------------- per-layer bias term: b3 @ h.sum(1)
__global__ void __launch_bounds__(64) k_prelude(const float* __restrict__ h,
                                                const float* __restrict__ be3,
                                                float* __restrict__ biasmsg) {
  __shared__ float sh[64];
  int b = blockIdx.x, i = threadIdx.x;
  float s = 0.f;
  for (int w = 0; w < 128; ++w) s += h[((long)b * 128 + w) * 64 + i];
  sh[i] = s;
  __syncthreads();
  float bm = 0.f;
  for (int k = 0; k < 64; ++k) bm += be3[i * 64 + k] * sh[k];  // i plays role of o
  biasmsg[b * 64 + i] = bm;
}

// ----------------------------------------------------------------- GRU update
__device__ inline float dot64(const float* a, const float* __restrict__ b) {
  float s = 0.f;
#pragma unroll 8
  for (int i = 0; i < 64; ++i) s += a[i] * b[i];
  return s;
}

__global__ void __launch_bounds__(64) k_gru(
    const float* __restrict__ msg, const float* __restrict__ biasmsg,
    const float* __restrict__ h, const float* __restrict__ Wih,
    const float* __restrict__ Whh, const float* __restrict__ bih,
    const float* __restrict__ bhh, const int* __restrict__ g_size,
    float* __restrict__ h_out, __bf16* __restrict__ hb_out) {
  __shared__ float sm[64], sh[64];
  int row = blockIdx.x, o = threadIdx.x;
  int b = row >> 7, v = row & 127;
  sm[o] = msg[(long)row * 64 + o] + biasmsg[b * 64 + o];
  sh[o] = h[(long)row * 64 + o];
  __syncthreads();
  float gir = bih[o]       + dot64(sm, Wih + (long)o * 64);
  float giz = bih[64 + o]  + dot64(sm, Wih + (long)(64 + o) * 64);
  float gin = bih[128 + o] + dot64(sm, Wih + (long)(128 + o) * 64);
  float ghr = bhh[o]       + dot64(sh, Whh + (long)o * 64);
  float ghz = bhh[64 + o]  + dot64(sh, Whh + (long)(64 + o) * 64);
  float ghn = bhh[128 + o] + dot64(sh, Whh + (long)(128 + o) * 64);
  float r = 1.f / (1.f + __expf(-(gir + ghr)));
  float z = 1.f / (1.f + __expf(-(giz + ghz)));
  float n = tanhf(gin + r * ghn);
  float hn = (1.f - z) * n + z * sh[o];
  if (v >= g_size[b]) hn = 0.f;
  h_out[(long)row * 64 + o] = hn;
  hb_out[(long)row * 64 + o] = f2bf(hn);
}

// ---------------------------------------- masked gate*val reduce + log_softmax
__global__ void __launch_bounds__(32) k_final(const float* __restrict__ gate,
                                              const float* __restrict__ val,
                                              const int* __restrict__ g_size,
                                              float* __restrict__ out) {
  int b = blockIdx.x, t = threadIdx.x;  // T = 32 = one wave
  int gs = g_size[b];
  if (gs > 128) gs = 128;
  if (gs < 0) gs = 0;
  float s = 0.f;
  for (int v = 0; v < gs; ++v) {
    long r = ((long)b * 128 + v) * 32 + t;
    s += gate[r] * val[r];
  }
  float mx = s;
  for (int off = 16; off > 0; off >>= 1) mx = fmaxf(mx, __shfl_xor(mx, off));
  float e = __expf(s - mx), se = e;
  for (int off = 16; off > 0; off >>= 1) se += __shfl_xor(se, off);
  out[b * 32 + t] = s - mx - __logf(se);
}

// ------------------------------------------------------------------- launcher
extern "C" void kernel_launch(void* const* d_in, const int* in_sizes, int n_in,
                              void* d_out, int out_size, void* d_ws, size_t ws_size,
                              hipStream_t stream) {
  const float* h_in = (const float*)d_in[0];
  const float* am   = (const float*)d_in[1];
  const int* g_size = (const int*)d_in[2];
  const float* We0 = (const float*)d_in[3];  const float* be0 = (const float*)d_in[4];
  const float* We1 = (const float*)d_in[5];  const float* be1 = (const float*)d_in[6];
  const float* We2 = (const float*)d_in[7];  const float* be2 = (const float*)d_in[8];
  const float* We3 = (const float*)d_in[9];  const float* be3 = (const float*)d_in[10];
  const float* Wih = (const float*)d_in[11]; const float* Whh = (const float*)d_in[12];
  const float* bih = (const float*)d_in[13]; const float* bhh = (const float*)d_in[14];
  // interleaved per reference: Wi{i}, bi{i}, Wj{i}, bj{i}
  const float* Wi[4], *bi[4], *Wj[4], *bj[4];
  for (int i = 0; i < 4; ++i) {
    Wi[i] = (const float*)d_in[15 + 4 * i + 0];
    bi[i] = (const float*)d_in[15 + 4 * i + 1];
    Wj[i] = (const float*)d_in[15 + 4 * i + 2];
    bj[i] = (const float*)d_in[15 + 4 * i + 3];
  }

  char* base = (char*)d_ws;
  size_t off = 0;
  auto alloc = [&](size_t bytes) -> void* {
    void* p = base + off;
    off = (off + bytes + 255) & ~(size_t)255;
    return p;
  };
  __bf16* f_bf  = (__bf16*)alloc(2UL * 262144 * 128);     // [B*N*N,128]
  __bf16* gT    = (__bf16*)alloc(2UL * 16 * 64 * 16384);  // [B,64,16384]
  __bf16* we3b  = (__bf16*)alloc(2UL * 128 * 4096);       // = WT for G-GEMM
  __bf16* we1T  = (__bf16*)alloc(2UL * 256 * 128);
  __bf16* we2T  = (__bf16*)alloc(2UL * 128 * 256);
  __bf16* wiT[4]; __bf16* wjT[4];
  wiT[0] = (__bf16*)alloc(2UL * 128 * 128);
  wiT[1] = (__bf16*)alloc(2UL * 256 * 128);
  wiT[2] = (__bf16*)alloc(2UL * 128 * 256);
  wiT[3] = (__bf16*)alloc(2UL * 32 * 128);
  wjT[0] = (__bf16*)alloc(2UL * 128 * 64);
  wjT[1] = (__bf16*)alloc(2UL * 256 * 128);
  wjT[2] = (__bf16*)alloc(2UL * 128 * 256);
  wjT[3] = (__bf16*)alloc(2UL * 32 * 128);
  float* hA      = (float*)alloc(4UL * 2048 * 64);
  float* hB      = (float*)alloc(4UL * 2048 * 64);
  __bf16* hbf    = (__bf16*)alloc(2UL * 2048 * 64);
  float* msg     = (float*)alloc(4UL * 2048 * 64);
  float* part    = (float*)alloc(4UL * 8 * 2048 * 64);    // K-split partials
  float* biasmsg = (float*)alloc(4UL * 16 * 64);
  __bf16* xg     = (__bf16*)alloc(2UL * 2048 * 128);
  __bf16* mA     = (__bf16*)alloc(2UL * 2048 * 256);
  __bf16* mB     = (__bf16*)alloc(2UL * 2048 * 256);
  float* gate    = (float*)alloc(4UL * 2048 * 32);
  float* val     = (float*)alloc(4UL * 2048 * 32);
  (void)in_sizes; (void)n_in; (void)out_size; (void)ws_size;

  auto cdiv = [](int a, int b) { return (a + b - 1) / b; };

  // weight prep (f32 -> bf16, transposed to [out][in], K zero-padded)
  k_transpose<<<cdiv(256 * 128, 256), 256, 0, stream>>>(We1, we1T, 128, 256, 128);
  k_transpose<<<cdiv(128 * 256, 256), 256, 0, stream>>>(We2, we2T, 256, 128, 256);
  k_convert<<<cdiv(128 * 4096, 256), 256, 0, stream>>>(We3, we3b, 128 * 4096);
  k_transpose<<<cdiv(128 * 128, 256), 256, 0, stream>>>(Wi[0], wiT[0], 96, 128, 128);
  k_transpose<<<cdiv(256 * 128, 256), 256, 0, stream>>>(Wi[1], wiT[1], 128, 256, 128);
  k_transpose<<<cdiv(128 * 256, 256), 256, 0, stream>>>(Wi[2], wiT[2], 256, 128, 256);
  k_transpose<<<cdiv(32 * 128, 256), 256, 0, stream>>>(Wi[3], wiT[3], 128, 32, 128);
  k_transpose<<<cdiv(128 * 64, 256), 256, 0, stream>>>(Wj[0], wjT[0], 64, 128, 64);
  k_transpose<<<cdiv(256 * 128, 256), 256, 0, stream>>>(Wj[1], wjT[1], 128, 256, 128);
  k_transpose<<<cdiv(128 * 256, 256), 256, 0, stream>>>(Wj[2], wjT[2], 256, 128, 256);
  k_transpose<<<cdiv(32 * 128, 256), 256, 0, stream>>>(Wj[3], wjT[3], 128, 32, 128);
  k_h0<<<cdiv(2048 * 64, 256), 256, 0, stream>>>(h_in, hA, hbf);

  // edge network (once): f = relu-MLP(am), stored bf16
  k_edge<<<4096, 128, 0, stream>>>(am, We0, be0, we1T, be1, we2T, be2, f_bf);

  // 3 weight-tied GGNN layers
  float* hc = hA;
  float* hn = hB;
  for (int l = 0; l < 3; ++l) {
    k_prelude<<<16, 64, 0, stream>>>(hc, be3, biasmsg);
    // G^T[b,o,w*128+k] = sum_i h[b,w,i] * We3[k, o*64+i]
    k_gemm<<<dim3(128, 128, 1), 32, 0, stream>>>(
        hbf, 0L, 64, we3b, 0L, 64, nullptr,
        64, 64, 2048, 8192, 0, 2, gT, 0);
    // msg[b,v,o] = sum_{w,k} f[b,v,w*128+k] * G^T[b,o,w*128+k]  (K split by 8)
    k_gemm<<<dim3(128, 1, 8), 32, 0, stream>>>(
        f_bf, (long)128 * 16384, 16384, gT, (long)64 * 16384, 16384, nullptr,
        16384, 2048, 128, 64, 0, 0, part, 64);
    k_reduce<<<cdiv(2048 * 64, 256), 256, 0, stream>>>(part, msg, 2048 * 64, 8);
    k_gru<<<2048, 64, 0, stream>>>(msg, biasmsg, hc, Wih, Whh, bih, bhh, g_size,
                                   hn, hbf);
    float* t = hc; hc = hn; hn = t;
  }

  // readout
  k_xg<<<cdiv(2048 * 128, 256), 256, 0, stream>>>(hbf, h_in, xg);
  // gate = sigmoid(MLP_i([h, h_in]))
  k_gemm<<<dim3(128, 2, 1), 32, 0, stream>>>(xg, 0L, 128, wiT[0], 0L, 128, bi[0],
                                             128, 128, 2048, 128, 1, 1, mA, 128);
  k_gemm<<<dim3(128, 4, 1), 32, 0, stream>>>(mA, 0L, 128, wiT[1], 0L, 128, bi[1],
                                             128, 128, 2048, 256, 1, 1, mB, 256);
  k_gemm<<<dim3(128, 2, 1), 32, 0, stream>>>(mB, 0L, 256, wiT[2], 0L, 256, bi[2],
                                             256, 256, 2048, 128, 1, 1, mA, 128);
  k_gemm<<<dim3(128, 1, 1), 32, 0, stream>>>(mA, 0L, 128, wiT[3], 0L, 128, bi[3],
                                             128, 128, 2048, 32, 2, 0, gate, 32);
  // val = MLP_j(h)
  k_gemm<<<dim3(128, 2, 1), 32, 0, stream>>>(hbf, 0L, 64, wjT[0], 0L, 64, bj[0],
                                             64, 64, 2048, 128, 1, 1, mB, 128);
  k_gemm<<<dim3(128, 4, 1), 32, 0, stream>>>(mB, 0L, 128, wjT[1], 0L, 128, bj[1],
                                             128, 128, 2048, 256, 1, 1, mA, 256);
  k_gemm<<<dim3(128, 2, 1), 32, 0, stream>>>(mA, 0L, 256, wjT[2], 0L, 256, bj[2],
                                             256, 256, 2048, 128, 1, 1, mB, 128);
  k_gemm<<<dim3(128, 1, 1), 32, 0, stream>>>(mB, 0L, 128, wjT[3], 0L, 128, bj[3],
                                             128, 128, 2048, 32, 0, 0, val, 32);

  k_final<<<16, 32, 0, stream>>>(gate, val, g_size, (float*)d_out);
}
